// SemiSupervisedLoss_78632261255934
// MI455X (gfx1250) — matrix operand order repaired and checked by reading, not compiled, not run on hardware
//
#include <hip/hip_runtime.h>
#include <math.h>

typedef __attribute__((ext_vector_type(16))) _Float16 v16h;
typedef __attribute__((ext_vector_type(8)))  float    v8f;

#define ALPHA_W   0.7f
#define BETA_W    0.3f
#define TEMP_INV  10.0f      // 1/0.1
#define CONF_THR  0.8f
#define NUM_NEG   3
#define MAX_EDGES_SAMPLED 1000
#define HIDDEN    128
#define NROWS     (16 * (2 + NUM_NEG))   // 80 staged embedding rows per tile
#define ROW_F     132                    // 128 + 4 pad floats: conflict-free LDS banks

// acc layout in d_ws (8 floats):
// 0: sup ce sum   1: sup count
// 2: cons ce sum  3: cons count
// 4: prop weighted diff sum  5: prop cross count
// 6: pos loss sum 7: neg loss sum

__device__ __forceinline__ float log_sigmoid_f(float x) {
    return fminf(x, 0.0f) - log1pf(expf(-fabsf(x)));   // stable log sigmoid
}

__device__ __forceinline__ unsigned hash_u32(unsigned x) {
    x ^= x >> 16; x *= 0x7feb352dU;
    x ^= x >> 15; x *= 0x846ca68bU;
    x ^= x >> 16;
    return x;
}

__device__ __forceinline__ float wave_sum(float v) {
#pragma unroll
    for (int off = 16; off > 0; off >>= 1) v += __shfl_down(v, off, 32);
    return v;
}

__global__ void init_acc_kernel(float* __restrict__ acc) {
    int t = threadIdx.x;
    if (t < 8) acc[t] = 0.0f;
}

// -------- Node pass: supervised CE + consistency CE (streaming reduction) ----
__global__ __launch_bounds__(256)
void node_pass_kernel(const float* __restrict__ logits,
                      const int*   __restrict__ labels,
                      const unsigned char* __restrict__ lab_mask,
                      float* __restrict__ acc, int n) {
    __shared__ float s[4];
    int tid  = threadIdx.x;
    int lane = tid & 31;
    if (tid < 4) s[tid] = 0.0f;
    __syncthreads();

    int i = blockIdx.x * blockDim.x + tid;
    float supce = 0.f, supcnt = 0.f, conce = 0.f, concnt = 0.f;
    if (i < n) {
        float2 l  = ((const float2*)logits)[i];
        bool  lab = lab_mask[i] != 0;
        float mx  = fmaxf(l.x, l.y);
        float lse = mx + logf(expf(l.x - mx) + expf(l.y - mx));
        if (lab) {
            int lb = labels[i];
            supce  = lse - (lb ? l.y : l.x);
            supcnt = 1.0f;
        }
        // max softmax prob at T=0.1; unlabeled == !labeled by construction
        float conf = 1.0f / (1.0f + expf(-fabsf(l.x - l.y) * TEMP_INV));
        if (!lab && (conf > CONF_THR)) {
            conce  = lse - mx;            // CE(logits, argmax) = lse - max logit
            concnt = 1.0f;
        }
    }
    supce = wave_sum(supce); supcnt = wave_sum(supcnt);
    conce = wave_sum(conce); concnt = wave_sum(concnt);
    if (lane == 0) {
        atomicAdd(&s[0], supce); atomicAdd(&s[1], supcnt);
        atomicAdd(&s[2], conce); atomicAdd(&s[3], concnt);
    }
    __syncthreads();
    if (tid < 4) atomicAdd(&acc[tid], s[tid]);
}

// -------- Edge pass: propagation loss over all 4M edges (dominant phase) ----
__global__ __launch_bounds__(256)
void edge_pass_kernel(const float* __restrict__ logits,
                      const int*   __restrict__ edge_index,
                      const float* __restrict__ ew,
                      const unsigned char* __restrict__ lab_mask,
                      float* __restrict__ acc, int nE) {
    __shared__ float s[2];
    int tid  = threadIdx.x;
    int lane = tid & 31;
    if (tid < 2) s[tid] = 0.0f;
    __syncthreads();

    int e = blockIdx.x * blockDim.x + tid;
    float dsum = 0.f, csum = 0.f;
    if (e < nE) {
        if (e + 16384 < nE) {
            __builtin_prefetch(&edge_index[e + 16384], 0, 1);       // global_prefetch_b8
            __builtin_prefetch(&edge_index[nE + e + 16384], 0, 1);
        }
        int sN = edge_index[e];
        int dN = edge_index[nE + e];
        // unlabeled == ~labeled  =>  cross-edge iff labels differ
        if ((lab_mask[sN] != 0) != (lab_mask[dN] != 0)) {
            float2 ls = ((const float2*)logits)[sN];   // 4MB table: L2-resident
            float2 ld = ((const float2*)logits)[dN];
            float p0s = 1.0f / (1.0f + expf(ls.y - ls.x));
            float p0d = 1.0f / (1.0f + expf(ld.y - ld.x));
            float dd  = p0s - p0d;
            dsum = 2.0f * dd * dd * ew[e];             // p1 = 1-p0 doubles the diff
            csum = 1.0f;
        }
    }
    dsum = wave_sum(dsum);
    csum = wave_sum(csum);
    if (lane == 0) { atomicAdd(&s[0], dsum); atomicAdd(&s[1], csum); }
    __syncthreads();
    if (tid == 0) atomicAdd(&acc[4], s[0]);
    if (tid == 1) atomicAdd(&acc[5], s[1]);
}

// -------- Graph (skip-gram) pass: async-staged LDS + WMMA dot products ------
// One wave32 per tile of 16 sampled edges. 80 embedding rows are staged
// global->LDS with async copies (one b128/lane = a full 512B row per
// instruction), then A = 16x32 f16 src slices / B = 32x16 f16 dst/neg slices
// feed v_wmma_f32_16x16x32_f16; the C diagonal yields the 16 dot products.
__global__ __launch_bounds__(32)
void graph_pass_kernel(const float* __restrict__ emb,
                       const int*   __restrict__ edge_index,
                       const float* __restrict__ ew,
                       float* __restrict__ acc, int nE) {
    __shared__ int   nodesLDS[NROWS];
    __shared__ float rows[NROWS * ROW_F];

    const int  lane = threadIdx.x;
    const int  col  = lane & 15;
    const bool lo   = lane < 16;

    // Deterministic "random" edge sample for slot j (stand-in for JAX RNG)
    int j = blockIdx.x * 16 + col;
    bool valid    = j < MAX_EDGES_SAMPLED;
    unsigned h    = hash_u32((unsigned)j * 0x9E3779B9u + 0x85ebca6bu);
    unsigned eidx = h % (unsigned)nE;
    int srcN = edge_index[eidx];
    int dstN = edge_index[nE + (int)eidx];
    float w  = valid ? ew[eidx] : 0.0f;

    if (lo) {
        nodesLDS[5 * col + 0] = srcN;
        nodesLDS[5 * col + 1] = dstN;
#pragma unroll
        for (int q = 0; q < NUM_NEG; ++q) {
            // negative ~ endpoint of a hashed random edge (deg-proportional)
            unsigned hq = hash_u32(h ^ (0x01000193u * (unsigned)(q + 1)));
            unsigned re = hq % (unsigned)nE;
            nodesLDS[5 * col + 2 + q] = edge_index[((hq >> 31) ? nE : 0) + (int)re];
        }
    }
    __syncthreads();

    // Stage all 80 rows global -> LDS via the async copy engine.
    for (int r = 0; r < NROWS; ++r) {
        int node = nodesLDS[r];
        const float* g  = emb + (size_t)node * HIDDEN + lane * 4;
        unsigned ldsOff = (unsigned)(size_t)(const void*)&rows[r * ROW_F + lane * 4];
        asm volatile("global_load_async_to_lds_b128 %0, %1, off"
                     :: "v"(ldsOff), "v"(g) : "memory");
    }
    asm volatile("s_wait_asynccnt 0x0" ::: "memory");
    __syncthreads();

    const float* aR  = &rows[(5 * col + 0) * ROW_F];   // A row M = lane&15
    const float* bpR = &rows[(5 * col + 1) * ROW_F];   // B col N = lane&15
    const float* b0R = &rows[(5 * col + 2) * ROW_F];
    const float* b1R = &rows[(5 * col + 3) * ROW_F];
    const float* b2R = &rows[(5 * col + 4) * ROW_F];

    const int koffA = lo ? 0 : 16;   // contiguous k = 16h + s (matches B)
    const int koffB = lo ? 0 : 16;   // contiguous k = 16h + s (matches A)

    v8f cpos = {}; v8f cn0 = {}; v8f cn1 = {}; v8f cn2 = {};

    for (int kb = 0; kb < HIDDEN; kb += 32) {
        float4 ar[4], bpr[4], b0r[4], b1r[4], b2r[4];
        ar[0] = *(const float4*)(aR + kb + koffA);
        ar[1] = *(const float4*)(aR + kb + koffA + 4);
        ar[2] = *(const float4*)(aR + kb + koffA + 8);
        ar[3] = *(const float4*)(aR + kb + koffA + 12);
#pragma unroll
        for (int c = 0; c < 4; ++c) {
            bpr[c] = *(const float4*)(bpR + kb + koffB + 4 * c);
            b0r[c] = *(const float4*)(b0R + kb + koffB + 4 * c);
            b1r[c] = *(const float4*)(b1R + kb + koffB + 4 * c);
            b2r[c] = *(const float4*)(b2R + kb + koffB + 4 * c);
        }
        v16h a, bp, b0, b1, b2;
#pragma unroll
        for (int c = 0; c < 4; ++c) {
            a[4*c+0]  = (_Float16)ar[c].x;  a[4*c+1]  = (_Float16)ar[c].y;
            a[4*c+2]  = (_Float16)ar[c].z;  a[4*c+3]  = (_Float16)ar[c].w;
            bp[4*c+0] = (_Float16)bpr[c].x; bp[4*c+1] = (_Float16)bpr[c].y;
            bp[4*c+2] = (_Float16)bpr[c].z; bp[4*c+3] = (_Float16)bpr[c].w;
            b0[4*c+0] = (_Float16)b0r[c].x; b0[4*c+1] = (_Float16)b0r[c].y;
            b0[4*c+2] = (_Float16)b0r[c].z; b0[4*c+3] = (_Float16)b0r[c].w;
            b1[4*c+0] = (_Float16)b1r[c].x; b1[4*c+1] = (_Float16)b1r[c].y;
            b1[4*c+2] = (_Float16)b1r[c].z; b1[4*c+3] = (_Float16)b1r[c].w;
            b2[4*c+0] = (_Float16)b2r[c].x; b2[4*c+1] = (_Float16)b2r[c].y;
            b2[4*c+2] = (_Float16)b2r[c].z; b2[4*c+3] = (_Float16)b2r[c].w;
        }
        // D = A x B + C ; A tile reused across pos + 3 negatives
        cpos = __builtin_amdgcn_wmma_f32_16x16x32_f16(false, a, false, bp, (short)0, cpos, false, false);
        cn0  = __builtin_amdgcn_wmma_f32_16x16x32_f16(false, a, false, b0, (short)0, cn0,  false, false);
        cn1  = __builtin_amdgcn_wmma_f32_16x16x32_f16(false, a, false, b1, (short)0, cn1,  false, false);
        cn2  = __builtin_amdgcn_wmma_f32_16x16x32_f16(false, a, false, b2, (short)0, cn2,  false, false);
    }

    // Diagonal extraction: C VGPR v -> (M=v, N=lane) lanes 0-15; (M=v+8, N=lane-16).
    // Edge r(<8): vgpr r, lane r.  Edge r(>=8): vgpr r-8, lane 24+(r-8).
    float sp = 0.f, s0 = 0.f, s1 = 0.f, s2 = 0.f;
#pragma unroll
    for (int v = 0; v < 8; ++v) {
        bool pick = (lane == v) || (lane == 24 + v);
        sp = pick ? cpos[v] : sp;
        s0 = pick ? cn0[v]  : s0;
        s1 = pick ? cn1[v]  : s1;
        s2 = pick ? cn2[v]  : s2;
    }
    bool diagLane = (lane < 8) || (lane >= 24);
    // diag lane's own (w) matches its diagonal edge: col = lane & 15.
    float posc = 0.f, negc = 0.f;
    if (diagLane) {
        posc = -log_sigmoid_f(sp) * w;
        negc = (-log_sigmoid_f(-s0) - log_sigmoid_f(-s1) - log_sigmoid_f(-s2)) * w;
    }
    posc = wave_sum(posc);
    negc = wave_sum(negc);
    if (lane == 0) {
        atomicAdd(&acc[6], posc);
        atomicAdd(&acc[7], negc);
    }
}

// -------- Finalize: combine the 5 losses -------------------------------------
__global__ void finalize_kernel(const float* __restrict__ acc,
                                float* __restrict__ out) {
    if (threadIdx.x == 0 && blockIdx.x == 0) {
        float sup  = (acc[1] > 0.f) ? acc[0] / fmaxf(acc[1], 1.f) : 0.f;
        float cons = (acc[3] > 0.f) ? acc[2] / fmaxf(acc[3], 1.f) : 0.f;
        float prop = acc[4] / fmaxf(acc[5], 1.f);
        float graph = acc[6] / (float)MAX_EDGES_SAMPLED
                    + acc[7] / (float)(MAX_EDGES_SAMPLED * NUM_NEG);
        float total = ALPHA_W * sup + BETA_W * graph + 0.1f * cons + 0.2f * prop;
        out[0] = total; out[1] = sup; out[2] = graph; out[3] = cons; out[4] = prop;
    }
}

extern "C" void kernel_launch(void* const* d_in, const int* in_sizes, int n_in,
                              void* d_out, int out_size, void* d_ws, size_t ws_size,
                              hipStream_t stream) {
    (void)n_in; (void)out_size; (void)ws_size;
    const float*         logits = (const float*)d_in[0];
    const int*           labels = (const int*)d_in[1];
    const float*         emb    = (const float*)d_in[2];
    const int*           eidx   = (const int*)d_in[3];
    const float*         ew     = (const float*)d_in[4];
    const unsigned char* lab    = (const unsigned char*)d_in[5];

    int n  = in_sizes[0] / 2;   // N_NODES
    int nE = in_sizes[4];       // N_EDGES

    float* acc = (float*)d_ws;

    init_acc_kernel<<<1, 32, 0, stream>>>(acc);
    node_pass_kernel<<<(n + 255) / 256, 256, 0, stream>>>(logits, labels, lab, acc, n);
    edge_pass_kernel<<<(nE + 255) / 256, 256, 0, stream>>>(logits, eidx, ew, lab, acc, nE);
    graph_pass_kernel<<<(MAX_EDGES_SAMPLED + 15) / 16, 32, 0, stream>>>(emb, eidx, ew, acc, nE);
    finalize_kernel<<<1, 1, 0, stream>>>(acc, (float*)d_out);
}
